// SelfAttention_22273700397706
// MI455X (gfx1250) — matrix operand order repaired and checked by reading, not compiled, not run on hardware
//
#include <hip/hip_runtime.h>
#include <hip/hip_bf16.h>

// ---------------------------------------------------------------------------
// Self-attention forward for MI455X (gfx1250, wave32, WMMA).
// N=4, L=2048, E=1024, H=16, D=64.
// All matmuls via v_wmma_f32_16x16x32_bf16 (f32 accumulate).
// Flash-style single-pass attention (online softmax): the 1 GB L x L score
// matrix is never materialized; only the attention mask (~1.07 GB int32) is
// streamed once, which is the bandwidth floor (~46 us at 23.3 TB/s).
// Softmax row reductions use DPP row_ror rotate-reduce (VALU pipe) instead of
// ds_bpermute so they don't contend with LDS fragment loads.
// ---------------------------------------------------------------------------

typedef __bf16 bf16;
typedef __attribute__((ext_vector_type(16))) __bf16 v16bf;
typedef __attribute__((ext_vector_type(8)))  __bf16 bf16x8;
typedef __attribute__((ext_vector_type(8)))  float   v8f;

#define NB  4
#define LQ  2048
#define EMB 1024
#define NH  16
#define HD  64

__device__ __forceinline__ bf16 f2bf(float x) {
  union { float f; unsigned u; } v; v.f = x;
  unsigned r = v.u + 0x7FFFu + ((v.u >> 16) & 1u);   // round-to-nearest-even
  unsigned short hs = (unsigned short)(r >> 16);
  union { unsigned short s; bf16 b; } o; o.s = hs; return o.b;
}

__device__ __forceinline__ v8f zero8() {
  v8f z;
#pragma unroll
  for (int i = 0; i < 8; ++i) z[i] = 0.0f;
  return z;
}

// ---- 16-lane-row rotate-reduce via DPP (row_ror stays inside 16-lane rows,
// matching the WMMA C-fragment row mapping m = 8*(lane>>4) + r). -------------
#if __has_builtin(__builtin_amdgcn_update_dpp)
#define DPP_ROR_F(x, n)                                                      \
  __int_as_float(__builtin_amdgcn_update_dpp(                                \
      0, __float_as_int(x), 0x120 | (n), 0xf, 0xf, 1))
__device__ __forceinline__ float row_reduce_max(float x) {
  x = fmaxf(x, DPP_ROR_F(x, 8));
  x = fmaxf(x, DPP_ROR_F(x, 4));
  x = fmaxf(x, DPP_ROR_F(x, 2));
  x = fmaxf(x, DPP_ROR_F(x, 1));
  return x;
}
__device__ __forceinline__ float row_reduce_sum(float x) {
  x += DPP_ROR_F(x, 8);
  x += DPP_ROR_F(x, 4);
  x += DPP_ROR_F(x, 2);
  x += DPP_ROR_F(x, 1);
  return x;
}
#else
__device__ __forceinline__ float row_reduce_max(float x) {
#pragma unroll
  for (int off = 1; off < 16; off <<= 1) x = fmaxf(x, __shfl_xor(x, off, 32));
  return x;
}
__device__ __forceinline__ float row_reduce_sum(float x) {
#pragma unroll
  for (int off = 1; off < 16; off <<= 1) x += __shfl_xor(x, off, 32);
  return x;
}
#endif

union FragU { v16bf v; bf16x8 h[2]; };

// A-matrix 16x32 bf16 fragment: lane half=lane>>4, row m=lane&15.
// element e -> k = (e/8)*16 + half*8 + (e%8)  => chunks at [half*8], [16+half*8]
__device__ __forceinline__ v16bf load_frag_A(const bf16* rowptr, int half) {
  FragU f;
  f.h[0] = *reinterpret_cast<const bf16x8*>(rowptr + half * 8);
  f.h[1] = *reinterpret_cast<const bf16x8*>(rowptr + 16 + half * 8);
  return f.v;
}
// B-matrix 32x16 bf16 fragment read from a K-contiguous (transposed) row:
// lane col n=lane&15, element e -> k = (e/8)*16 + half*8 + (e%8)  => chunks at [half*8], [16+half*8]
__device__ __forceinline__ v16bf load_frag_B(const bf16* rowptr, int half) {
  FragU f;
  f.h[0] = *reinterpret_cast<const bf16x8*>(rowptr + half * 8);
  f.h[1] = *reinterpret_cast<const bf16x8*>(rowptr + 16 + half * 8);
  return f.v;
}

__device__ __forceinline__ v8f wmma_bf16(v16bf a, v16bf b, v8f c) {
  return __builtin_amdgcn_wmma_f32_16x16x32_bf16(
      /*neg_a=*/false, a, /*neg_b=*/false, b,
      /*c_mod=*/(short)0, c, /*reuse_a=*/false, /*reuse_b=*/false);
}

// ---------------------------------------------------------------------------
// Kernel 1: f32 -> bf16 conversion
// ---------------------------------------------------------------------------
__global__ void cvt_f32_bf16(const float* __restrict__ src,
                             bf16* __restrict__ dst, int n) {
  int i = blockIdx.x * blockDim.x + threadIdx.x;
  int stride = gridDim.x * blockDim.x;
  for (; i < n; i += stride) dst[i] = f2bf(src[i]);
}

// ---------------------------------------------------------------------------
// Kernel 2/4: 128x128-tile bf16 GEMM, k-tile 64, f32 accumulate.
// MODE 0: C = A*B + bias -> scatter bf16 into head-split Q/K/V [N,H,L,D]
// MODE 1: C = A*B + bias -> fp32 out (row-major [M, Ncols])
// Block: 256 threads = 8 waves as 2(m) x 4(n); each wave: 64x32 = 4x2 subtiles.
// ---------------------------------------------------------------------------
template <int MODE>
__global__ __launch_bounds__(256)
void gemm128(const bf16* __restrict__ A, const bf16* __restrict__ B,
             const float* __restrict__ bias, int M, int Ncols, int Kd,
             bf16* __restrict__ qo, bf16* __restrict__ ko,
             bf16* __restrict__ vo, float* __restrict__ fo) {
  __shared__ bf16 As[128][72];    // [m][k], k-stride padded 64->72
  __shared__ bf16 BsT[128][72];   // [n][k] (transposed at fill time)

  const int tid  = threadIdx.x;
  const int lane = tid & 31;
  const int w    = tid >> 5;
  const int wm   = (w & 1) * 64;   // wave row base within tile
  const int wn   = (w >> 1) * 32;  // wave col base within tile
  const int half = lane >> 4;
  const int ln   = lane & 15;
  const int bm   = blockIdx.y * 128;
  const int bn   = blockIdx.x * 128;

  v8f acc[4][2];
#pragma unroll
  for (int i = 0; i < 4; ++i)
#pragma unroll
    for (int j = 0; j < 2; ++j) acc[i][j] = zero8();

  for (int kt = 0; kt < Kd; kt += 64) {
    __syncthreads();
    // A tile: 128 rows x 64 k  (1024 segments of 8 bf16; 4 per thread)
#pragma unroll
    for (int it = 0; it < 4; ++it) {
      int idx = tid + it * 256;
      int row = idx >> 3;                // 0..127
      int seg = idx & 7;                 // 0..7  (k-offset seg*8)
      bf16x8 vd = *reinterpret_cast<const bf16x8*>(
          &A[(size_t)(bm + row) * Kd + kt + seg * 8]);
      *reinterpret_cast<bf16x8*>(&As[row][seg * 8]) = vd;
    }
    // B tile: 64 k x 128 n, transposed into [n][k]
#pragma unroll
    for (int it = 0; it < 4; ++it) {
      int idx = tid + it * 256;
      int kk  = idx >> 4;                // 0..63
      int c0  = (idx & 15) * 8;          // col segment
      bf16x8 vd = *reinterpret_cast<const bf16x8*>(
          &B[(size_t)(kt + kk) * Ncols + bn + c0]);
#pragma unroll
      for (int e = 0; e < 8; ++e) BsT[c0 + e][kk] = vd[e];
    }
    __syncthreads();

#pragma unroll
    for (int c = 0; c < 2; ++c) {        // two 32-wide k chunks
      v16bf bfrag[2];
#pragma unroll
      for (int j = 0; j < 2; ++j)
        bfrag[j] = load_frag_B(&BsT[wn + j * 16 + ln][c * 32], half);
#pragma unroll
      for (int i = 0; i < 4; ++i) {
        v16bf afrag = load_frag_A(&As[wm + i * 16 + ln][c * 32], half);
#pragma unroll
        for (int j = 0; j < 2; ++j)
          acc[i][j] = wmma_bf16(afrag, bfrag[j], acc[i][j]);
      }
    }
  }

  // Epilogue. C layout: element r -> row = half*8 + r, col = ln.
#pragma unroll
  for (int i = 0; i < 4; ++i) {
#pragma unroll
    for (int j = 0; j < 2; ++j) {
#pragma unroll
      for (int r = 0; r < 8; ++r) {
        int row = bm + wm + i * 16 + half * 8 + r;
        int col = bn + wn + j * 16 + ln;
        float vv = acc[i][j][r] + bias[col];
        if (MODE == 0) {
          int proj = col >> 10;              // 0=Q 1=K 2=V
          int e    = col & 1023;
          int h    = e >> 6;
          int d    = e & 63;
          int nb   = row >> 11;              // L = 2048
          int l    = row & 2047;
          size_t di = (((size_t)nb * NH + h) * LQ + l) * HD + d;
          bf16 bv = f2bf(vv);
          if (proj == 0)      qo[di] = bv;
          else if (proj == 1) ko[di] = bv;
          else                vo[di] = bv;
        } else {
          fo[(size_t)row * Ncols + col] = vv;
        }
      }
    }
  }
}

// ---------------------------------------------------------------------------
// Kernel 3: flash attention, 64-wide k tiles. grid = (L/64, N*H), block = 128.
// Each wave owns a 16-row q strip; waves share K/V tiles staged in LDS.
// ---------------------------------------------------------------------------
__global__ __launch_bounds__(128)
void attn_kernel(const bf16* __restrict__ Q, const bf16* __restrict__ K,
                 const bf16* __restrict__ V, const int* __restrict__ mask,
                 bf16* __restrict__ O) {
  __shared__ bf16 Qs[64][72];       // [q][d], A-layout rows
  __shared__ bf16 Ks[64][72];       // [k][d] -> B frags for Q*K^T read row n
  __shared__ bf16 VsT[64][72];      // [d][k] -> B frags for P*V
  __shared__ bf16 Ps[4][16][72];    // per-wave P staging (C -> A relayout)

  const int tid  = threadIdx.x;
  const int lane = tid & 31;
  const int w    = tid >> 5;        // wave 0..3, q-strip rows 16w..16w+15
  const int half = lane >> 4;
  const int ln   = lane & 15;
  const int qb   = blockIdx.x * 64;
  const int nh   = blockIdx.y;      // n*H + h
  const bf16* Qh = Q + (size_t)nh * LQ * HD;
  const bf16* Kh = K + (size_t)nh * LQ * HD;
  const bf16* Vh = V + (size_t)nh * LQ * HD;
  const int* maskh = mask + (size_t)nh * LQ * LQ;

  // Load Q block (64 x 64)
  for (int i = tid; i < 64 * 8; i += 128) {
    int row = i >> 3, seg = i & 7;
    *reinterpret_cast<bf16x8*>(&Qs[row][seg * 8]) =
        *reinterpret_cast<const bf16x8*>(&Qh[(size_t)(qb + row) * HD + seg * 8]);
  }
  __syncthreads();

  v16bf qa[2];
  qa[0] = load_frag_A(&Qs[w * 16 + ln][0], half);    // d 0..31
  qa[1] = load_frag_A(&Qs[w * 16 + ln][32], half);   // d 32..63

  v8f oacc[4];
#pragma unroll
  for (int d = 0; d < 4; ++d) oacc[d] = zero8();
  float mrun[8], lrun[8];
#pragma unroll
  for (int r = 0; r < 8; ++r) { mrun[r] = -__builtin_inff(); lrun[r] = 0.0f; }

  const float scale = 0.125f;   // D^-0.5

  for (int kt = 0; kt < LQ; kt += 64) {
    __syncthreads();   // prior iteration's frag reads done before refill
    for (int i = tid; i < 64 * 8; i += 128) {      // K tile 64x64
      int row = i >> 3, seg = i & 7;
      *reinterpret_cast<bf16x8*>(&Ks[row][seg * 8]) =
          *reinterpret_cast<const bf16x8*>(&Kh[(size_t)(kt + row) * HD + seg * 8]);
    }
    for (int i = tid; i < 64 * 8; i += 128) {      // V tile 64x64 -> transposed
      int row = i >> 3, seg = i & 7;
      bf16x8 vd = *reinterpret_cast<const bf16x8*>(
          &Vh[(size_t)(kt + row) * HD + seg * 8]);
#pragma unroll
      for (int e = 0; e < 8; ++e) VsT[seg * 8 + e][row] = vd[e];
    }
    __syncthreads();

    // S(16x64) = Q(16x64) * K^T(64x64): 4 col subtiles x 2 k chunks = 8 WMMA
    v8f s[4];
#pragma unroll
    for (int j = 0; j < 4; ++j) s[j] = zero8();
#pragma unroll
    for (int c = 0; c < 2; ++c) {
#pragma unroll
      for (int j = 0; j < 4; ++j) {
        v16bf bk = load_frag_B(&Ks[j * 16 + ln][c * 32], half);
        s[j] = wmma_bf16(qa[c], bk, s[j]);
      }
    }

    // mask + scale (mask streamed exactly once -> bandwidth floor)
    const int qrow_base = qb + w * 16 + half * 8;
#pragma unroll
    for (int r = 0; r < 8; ++r) {
      size_t mb = (size_t)(qrow_base + r) * LQ + kt;
#pragma unroll
      for (int j = 0; j < 4; ++j) {
        float v0 = s[j][r] * scale;
        if (maskh[mb + j * 16 + ln]) v0 = -__builtin_inff();
        s[j][r] = v0;
      }
    }

    // online softmax per row (row = 16 lanes of one half; DPP rotate-reduce)
#pragma unroll
    for (int r = 0; r < 8; ++r) {
      float mx = fmaxf(fmaxf(s[0][r], s[1][r]), fmaxf(s[2][r], s[3][r]));
      mx = row_reduce_max(mx);
      float mnew = fmaxf(fmaxf(mrun[r], mx), -1e30f);  // keep exp args finite
      float corr = __expf(mrun[r] - mnew);
      float p0 = __expf(s[0][r] - mnew);
      float p1 = __expf(s[1][r] - mnew);
      float p2 = __expf(s[2][r] - mnew);
      float p3 = __expf(s[3][r] - mnew);
      float ps = row_reduce_sum((p0 + p1) + (p2 + p3));
      lrun[r] = lrun[r] * corr + ps;
      mrun[r] = mnew;
#pragma unroll
      for (int d = 0; d < 4; ++d) oacc[d][r] *= corr;
      // stage P (C layout -> LDS row-major) for A-fragment reload
      Ps[w][half * 8 + r][ln]      = f2bf(p0);
      Ps[w][half * 8 + r][16 + ln] = f2bf(p1);
      Ps[w][half * 8 + r][32 + ln] = f2bf(p2);
      Ps[w][half * 8 + r][48 + ln] = f2bf(p3);
    }

    // O(16x64) += P(16x64) * V(64x64): 4 d subtiles x 2 k chunks = 8 WMMA
    v16bf pa[2];
    pa[0] = load_frag_A(&Ps[w][ln][0], half);
    pa[1] = load_frag_A(&Ps[w][ln][32], half);
#pragma unroll
    for (int d = 0; d < 4; ++d) {
#pragma unroll
      for (int c = 0; c < 2; ++c) {
        v16bf bv = load_frag_B(&VsT[d * 16 + ln][c * 32], half);
        oacc[d] = wmma_bf16(pa[c], bv, oacc[d]);
      }
    }
  }

  // Normalize and write O as [N, L, E] bf16 (re-merging heads)
  const int nb = nh >> 4, h = nh & 15;
#pragma unroll
  for (int d = 0; d < 4; ++d) {
#pragma unroll
    for (int r = 0; r < 8; ++r) {
      int l = qb + w * 16 + half * 8 + r;
      int e = h * HD + d * 16 + ln;
      float inv = (lrun[r] > 0.0f) ? (1.0f / lrun[r]) : 0.0f;
      O[((size_t)nb * LQ + l) * EMB + e] = f2bf(oacc[d][r] * inv);
    }
  }
}

// ---------------------------------------------------------------------------
// Host-side launch. Inputs: input, attn_mask, W_qkv, b_qkv, W_o, b_o.
// Workspace layout (bytes, all regions written before read):
//   [0)        inp_bf16   16,777,216
//   [16MB)     Wqkv_bf16   6,291,456
//   [+6MB)     Wo_bf16     2,097,152
//   then Q, K, V, O bf16 (16,777,216 each)  => ~88 MB total
// ---------------------------------------------------------------------------
extern "C" void kernel_launch(void* const* d_in, const int* in_sizes, int n_in,
                              void* d_out, int out_size, void* d_ws,
                              size_t ws_size, hipStream_t stream) {
  const float* input = (const float*)d_in[0];
  const int*   mask  = (const int*)d_in[1];
  const float* Wqkv  = (const float*)d_in[2];
  const float* bqkv  = (const float*)d_in[3];
  const float* Wo    = (const float*)d_in[4];
  const float* bo    = (const float*)d_in[5];
  float* out = (float*)d_out;

  char* ws = (char*)d_ws;
  bf16* inp_bf  = (bf16*)(ws);
  bf16* wqkv_bf = (bf16*)(ws + 16777216);
  bf16* wo_bf   = (bf16*)(ws + 23068672);
  bf16* Qb      = (bf16*)(ws + 25165824);
  bf16* Kb      = (bf16*)(ws + 41943040);
  bf16* Vb      = (bf16*)(ws + 58720256);
  bf16* Ob      = (bf16*)(ws + 75497472);

  cvt_f32_bf16<<<4096, 256, 0, stream>>>(input, inp_bf, NB * LQ * EMB);
  cvt_f32_bf16<<<4096, 256, 0, stream>>>(Wqkv, wqkv_bf, EMB * 3 * EMB);
  cvt_f32_bf16<<<2048, 256, 0, stream>>>(Wo, wo_bf, EMB * EMB);

  // QKV projection: [8192, 1024] x [1024, 3072]
  gemm128<0><<<dim3(24, 64), 256, 0, stream>>>(
      inp_bf, wqkv_bf, bqkv, NB * LQ, 3 * EMB, EMB, Qb, Kb, Vb, nullptr);

  // Flash attention over all (n, h) and 64-row q blocks
  attn_kernel<<<dim3(LQ / 64, NB * NH), 128, 0, stream>>>(Qb, Kb, Vb, mask, Ob);

  // Output projection: [8192, 1024] x [1024, 1024] -> fp32
  gemm128<1><<<dim3(8, 64), 256, 0, stream>>>(
      Ob, wo_bf, bo, NB * LQ, EMB, EMB, nullptr, nullptr, nullptr, out);
}
